// LnLstm_8710193676561
// MI455X (gfx1250) — compile-verified, matrix operand order checked by reading, not run on hardware
//
#include <hip/hip_runtime.h>

#define S_ 8
#define I_ 64
#define M_ 256

typedef _Float16 h16;
typedef __attribute__((ext_vector_type(16))) _Float16 v16h;
typedef __attribute__((ext_vector_type(8)))  float    v8f;

__device__ __forceinline__ v8f vzero() {
    v8f z = {0.f, 0.f, 0.f, 0.f, 0.f, 0.f, 0.f, 0.f};
    return z;
}

// K-offset pattern for the 16-bit 16x32 A/B WMMA fragment layout (ISA 7.12.2):
// VGPR v holds K = {base, base+1} with base = ((v&4)<<2) | ((v&3)<<1) | (g<<3)
__device__ __forceinline__ int kpat(int v, int g) {
    return ((v & 4) << 2) | ((v & 3) << 1) | (g << 3);
}

// Load one 16x32 f16 fragment from row-major storage. Caller passes the
// per-lane row pointer already offset by k0. Pairs are contiguous in K ->
// 8 dword loads (ds_load_b32 for LDS, global_load_b32 for global).
__device__ __forceinline__ v16h load_frag_h(const h16* base, int g) {
    union { v16h v; unsigned u[8]; } f;
#pragma unroll
    for (int v = 0; v < 8; ++v)
        f.u[v] = *reinterpret_cast<const unsigned*>(base + kpat(v, g));
    return f.v;
}

// Same, but source is f32 in memory (modulation): load float2, cvt+pack to 2 halves.
__device__ __forceinline__ v16h load_frag_f32(const float* base, int g) {
    union { v16h v; unsigned u[8]; } f;
#pragma unroll
    for (int v = 0; v < 8; ++v) {
        const float2 p = *reinterpret_cast<const float2*>(base + kpat(v, g));
        union { h16 h[2]; unsigned u; } c;
        c.h[0] = (h16)p.x;
        c.h[1] = (h16)p.y;
        f.u[v] = c.u;
    }
    return f.v;
}

__device__ __forceinline__ float fast_sigmoid(float x) { return 1.0f / (1.0f + __expf(-x)); }
__device__ __forceinline__ float fast_tanh(float x)    { return 1.0f - 2.0f / (1.0f + __expf(2.0f * x)); }

// ---------------------------------------------------------------------------
// b_eff[G][s,m] = b[G][s,m] + sum_k W[G][s,m,M+k] * h0[s,k]
// (h_prev is h0 broadcast across the batch -> the h-half of each gate GEMM is
//  a constant per (s,m); folding it halves the gate FLOPs exactly.)
// ---------------------------------------------------------------------------
__global__ void fold_bias_kernel(const float* __restrict__ Wi, const float* __restrict__ Wf,
                                 const float* __restrict__ Wg, const float* __restrict__ Wo,
                                 const float* __restrict__ bi, const float* __restrict__ bf,
                                 const float* __restrict__ bg, const float* __restrict__ bo,
                                 const float* __restrict__ h0, float* __restrict__ bEff)
{
    const int G  = blockIdx.y;
    const int sm = blockIdx.x * blockDim.x + threadIdx.x;
    if (sm >= S_ * M_) return;
    const float* W = (G == 0) ? Wi : (G == 1) ? Wf : (G == 2) ? Wg : Wo;
    const float* b = (G == 0) ? bi : (G == 1) ? bf : (G == 2) ? bg : bo;
    const int s = sm >> 8;
    float acc = b[sm];
    const float* wrow = W + (size_t)sm * (2 * M_) + M_;
    const float* hrow = h0 + s * M_;
    for (int k = 0; k < M_; ++k)
        acc += wrow[k] * hrow[k];
    bEff[G * (S_ * M_) + sm] = acc;
}

// ---------------------------------------------------------------------------
// Convert Wx (full) and the x-half (first M of 2M K-columns) of the four gate
// weights to f16, keeping [m][k] row-major (B-fragment friendly).
// ---------------------------------------------------------------------------
__global__ void convert_weights_kernel(const float* __restrict__ Wx,
                                       const float* __restrict__ Wi, const float* __restrict__ Wf,
                                       const float* __restrict__ Wg, const float* __restrict__ Wo,
                                       h16* __restrict__ Wx16, h16* __restrict__ Wg16)
{
    const int NX = S_ * M_ * I_;   // 131072
    const int NG = S_ * M_ * M_;   // 524288 per gate
    const int idx = blockIdx.x * blockDim.x + threadIdx.x;
    if (idx >= NX + 4 * NG) return;
    if (idx < NX) { Wx16[idx] = (h16)Wx[idx]; return; }
    const int t  = idx - NX;
    const int G  = t / NG;
    const int q  = t - G * NG;
    const int sm = q >> 8;          // s*256 + m
    const int k  = q & (M_ - 1);
    const float* W = (G == 0) ? Wi : (G == 1) ? Wf : (G == 2) ? Wg : Wo;
    Wg16[t] = (h16)W[(size_t)sm * (2 * M_) + k];
}

// ---------------------------------------------------------------------------
// Fused main kernel. Block = (stream s, 64 batch rows), 256 threads = 8 waves.
// Phase 1: x[64x256] = mod @ Wx^T + bx via WMMA, f16 tile into LDS.
// Phase 2: 4 gate GEMMs (K=256) from LDS x-tile, fused LSTM epilogue.
// Requires N % 64 == 0 (reference N = 16384).
// ---------------------------------------------------------------------------
__global__ __launch_bounds__(256) void lstm_main_kernel(
    const float* __restrict__ mod, const float* __restrict__ bx, const float* __restrict__ c0,
    const h16* __restrict__ Wx16, const h16* __restrict__ Wg16,
    const float* __restrict__ bEff, float* __restrict__ out)
{
    constexpr int XS_STRIDE = 268;               // padded (268 halves = 134 banks, conflict-dodging)
    __shared__ h16 xs[64 * XS_STRIDE];           // 34.3 KB

    const int s       = blockIdx.y;
    const int rowBase = blockIdx.x * 64;
    const int wave    = threadIdx.x >> 5;
    const int lane    = threadIdx.x & 31;
    const int g       = lane >> 4;
    const int lr      = lane & 15;

    // ---- Phase 1: x tile -------------------------------------------------
    {
        const int rt = wave >> 1;   // row tile 0..3 (16 rows each)
        const int ch = wave & 1;    // column half (128 cols each)
        v8f acc[8];
#pragma unroll
        for (int c = 0; c < 8; ++c) acc[c] = vzero();
#pragma unroll
        for (int kk = 0; kk < 2; ++kk) {
            const int k0 = kk * 32;
            const v16h a = load_frag_f32(
                mod + (size_t)(rowBase + rt * 16 + lr) * (S_ * I_) + s * I_ + k0, g);
#pragma unroll
            for (int c = 0; c < 8; ++c) {
                const v16h b = load_frag_h(
                    Wx16 + (size_t)(s * M_ + ch * 128 + c * 16 + lr) * I_ + k0, g);
                acc[c] = __builtin_amdgcn_wmma_f32_16x16x32_f16(
                    false, a, false, b, (short)0, acc[c], false, false);
            }
        }
#pragma unroll
        for (int c = 0; c < 8; ++c) {
            const int col = ch * 128 + c * 16 + lr;
            const float bias = bx[s * M_ + col];
#pragma unroll
            for (int r = 0; r < 8; ++r) {
                const int row = rt * 16 + r + 8 * g;
                xs[row * XS_STRIDE + col] = (h16)(acc[c][r] + bias);
            }
        }
    }
    __syncthreads();

    // ---- Phase 2: gate GEMMs + LSTM epilogue -----------------------------
    const int rp = wave & 1;    // row tiles {rp, rp+2}
    const int ct = wave >> 1;   // col tile 0..3 within chunk

    for (int chunk = 0; chunk < 4; ++chunk) {
        const int colT = chunk * 64 + ct * 16;
        v8f acc[4][2];
#pragma unroll
        for (int G = 0; G < 4; ++G) {
            acc[G][0] = vzero();
            acc[G][1] = vzero();
        }
#pragma unroll
        for (int kk = 0; kk < 8; ++kk) {
            const int k0 = kk * 32;
            const v16h a0 = load_frag_h(&xs[(rp * 16 + lr) * XS_STRIDE + k0], g);
            const v16h a1 = load_frag_h(&xs[((rp + 2) * 16 + lr) * XS_STRIDE + k0], g);
#pragma unroll
            for (int G = 0; G < 4; ++G) {
                const v16h b = load_frag_h(
                    Wg16 + (size_t)((G * S_ + s) * M_ + colT + lr) * M_ + k0, g);
                acc[G][0] = __builtin_amdgcn_wmma_f32_16x16x32_f16(
                    false, a0, false, b, (short)0, acc[G][0], false, false);
                acc[G][1] = __builtin_amdgcn_wmma_f32_16x16x32_f16(
                    false, a1, false, b, (short)0, acc[G][1], false, false);
            }
        }

        const int col = colT + lr;
        const float bI = bEff[0 * (S_ * M_) + s * M_ + col];
        const float bF = bEff[1 * (S_ * M_) + s * M_ + col];
        const float bG = bEff[2 * (S_ * M_) + s * M_ + col];
        const float bO = bEff[3 * (S_ * M_) + s * M_ + col];
        const float cp = c0[s * M_ + col];
#pragma unroll
        for (int t = 0; t < 2; ++t) {
#pragma unroll
            for (int r = 0; r < 8; ++r) {
                const int row = rowBase + (rp + 2 * t) * 16 + r + 8 * g;
                const float iv = fast_sigmoid(acc[0][t][r] + bI);
                const float fv = fast_sigmoid(acc[1][t][r] + bF);
                const float gv = fast_tanh   (acc[2][t][r] + bG);
                const float ov = fast_sigmoid(acc[3][t][r] + bO);
                const float cv = fv * cp + iv * gv;
                out[(size_t)row * (S_ * M_) + s * M_ + col] = ov * fast_tanh(cv);
            }
        }
    }
}

extern "C" void kernel_launch(void* const* d_in, const int* in_sizes, int n_in,
                              void* d_out, int out_size, void* d_ws, size_t ws_size,
                              hipStream_t stream)
{
    const float* mod = (const float*)d_in[0];
    const float* h0  = (const float*)d_in[1];
    const float* c0  = (const float*)d_in[2];
    const float* Wx  = (const float*)d_in[3];
    const float* bx  = (const float*)d_in[4];
    const float* Wi  = (const float*)d_in[5];
    const float* bi  = (const float*)d_in[6];
    const float* Wf  = (const float*)d_in[7];
    const float* bf  = (const float*)d_in[8];
    const float* Wg  = (const float*)d_in[9];
    const float* bg  = (const float*)d_in[10];
    const float* Wo  = (const float*)d_in[11];
    const float* bo  = (const float*)d_in[12];
    float* out = (float*)d_out;

    const int N = in_sizes[0] / (S_ * I_);   // 16384

    // Workspace layout: bEff (4*2048 f32) | Wx16 (131072 f16) | Wg16 (2097152 f16)
    float* bEff = (float*)d_ws;
    h16*   Wx16 = (h16*)(bEff + 4 * S_ * M_);
    h16*   Wg16 = Wx16 + (size_t)S_ * M_ * I_;

    fold_bias_kernel<<<dim3((S_ * M_ + 255) / 256, 4), 256, 0, stream>>>(
        Wi, Wf, Wg, Wo, bi, bf, bg, bo, h0, bEff);

    const int total = S_ * M_ * I_ + 4 * S_ * M_ * M_;
    convert_weights_kernel<<<(total + 255) / 256, 256, 0, stream>>>(
        Wx, Wi, Wf, Wg, Wo, Wx16, Wg16);

    lstm_main_kernel<<<dim3(N / 64, S_), 256, 0, stream>>>(
        mod, bx, c0, Wx16, Wg16, bEff, out);
}
